// Attention_83184926589559
// MI455X (gfx1250) — matrix operand order repaired and checked by reading, not compiled, not run on hardware
//
#include <hip/hip_runtime.h>
#include <stdint.h>

typedef __attribute__((ext_vector_type(16))) _Float16 v16h;
typedef __attribute__((ext_vector_type(8)))  float    v8f;
typedef __attribute__((ext_vector_type(4)))  unsigned int u32x4;
typedef __attribute__((ext_vector_type(8)))  int          i32x8;
typedef __attribute__((ext_vector_type(4)))  int          i32x4;

#define HEADS  8
#define DHEAD  64
#define NSEQ   4096
#define DMODEL 512
#define ATTN_SCALE 0.125f   // 64^-0.5

#if __has_builtin(__builtin_amdgcn_tensor_load_to_lds) && __has_builtin(__builtin_amdgcn_s_wait_tensorcnt)
#define USE_TDM 1
#else
#define USE_TDM 0
#endif

union FragU { v16h h; u32x4 q[2]; uint32_t u[8]; };

// A-fragment (16x32 f16, MxK), ISA 7.12.2. Lane l<16: M=l, K 0..7 (v0..3) /
// 16..23 (v4..7); lane l>=16: same M, K 8..15 / 24..31. base = (row0,k0) of a
// row-major f16 tile; rowStride in halves, MUST be a multiple of 8 (16B) so
// each half-fragment is one b128 load.
static __device__ __forceinline__ v16h load_a_frag(const _Float16* base, int rowStride, int lane) {
  FragU r;
  const int row = lane & 15;
  const int sel = (lane >> 4) & 1;
  const u32x4* p = (const u32x4*)(base + row * rowStride);
  r.q[0] = p[sel];        // K 0..7   or 8..15
  r.q[1] = p[2 + sel];    // K 16..23 or 24..31
  return r.h;
}

// B-fragment (32x16 f16, KxN). Lane l: N=l&15; K 0..7 / 16..23 (lanes 0-15) or
// 8..15 / 24..31 (lanes 16-31), 2 consecutive K per VGPR. base layout [n][k]
// (K-pairs contiguous per output column); rowStride multiple of 8 halves.
static __device__ __forceinline__ v16h load_b_frag(const _Float16* base, int rowStride, int lane) {
  FragU r;
  const int n   = lane & 15;
  const int sel = (lane >> 4) & 1;
  const u32x4* p = (const u32x4*)(base + n * rowStride);
  r.q[0] = p[sel];
  r.q[1] = p[2 + sel];
  return r.h;
}

static __device__ __forceinline__ v8f wmma_f16(v16h a, v16h b, v8f c) {
  return __builtin_amdgcn_wmma_f32_16x16x32_f16(false, a, false, b, (short)0, c, false, false);
}

// ---------------------------------------------------------------------------
// Kernel 1: fused QKV projection.  x[8192,512]f32 @ ([512,512]|[512,1024])f32
// -> q,k,v f16 in [B,h,N,d] (q pre-scaled). 4 waves, 64x64 tile, K-step 32.
// LDS row stride 40 halves = 80B: 16B-aligned AND conflict-free for b128
// (banks 20r mod 64 = {0,4,...,60}).
// ---------------------------------------------------------------------------
__global__ __launch_bounds__(128) void qkv_proj_kernel(
    const float* __restrict__ x, const float* __restrict__ Wq, const float* __restrict__ Wkv,
    _Float16* __restrict__ qh, _Float16* __restrict__ kh, _Float16* __restrict__ vh)
{
  __shared__ __align__(16) _Float16 As[64 * 40];   // [row][k]
  __shared__ __align__(16) _Float16 Bs[64 * 40];   // [col][k]
  const int tid = threadIdx.x, lane = tid & 31, wave = tid >> 5;
  const int rowTile = blockIdx.x * 64;
  const int gcBase  = blockIdx.y * 64;

  v8f acc[4] = {};

  for (int k0 = 0; k0 < DMODEL; k0 += 32) {
#pragma unroll
    for (int it = 0; it < 4; ++it) {                 // A: 512 float4, 4/thread
      int idx = tid + it * 128;
      int r  = idx >> 3;
      int c4 = (idx & 7) << 2;
      float4 f = *(const float4*)(x + (size_t)(rowTile + r) * DMODEL + k0 + c4);
      _Float16* d = As + r * 40 + c4;
      d[0] = (_Float16)f.x; d[1] = (_Float16)f.y; d[2] = (_Float16)f.z; d[3] = (_Float16)f.w;
    }
#pragma unroll
    for (int it = 0; it < 4; ++it) {                 // B: transpose into [col][k]
      int idx = tid + it * 128;
      int kr = idx >> 4;
      int c4 = (idx & 15) << 2;
      int gc = gcBase + c4;
      const float* src = (gc < 512) ? (Wq  + (size_t)(k0 + kr) * 512  + gc)
                                    : (Wkv + (size_t)(k0 + kr) * 1024 + (gc - 512));
      float4 f = *(const float4*)src;
      Bs[(c4 + 0) * 40 + kr] = (_Float16)f.x;
      Bs[(c4 + 1) * 40 + kr] = (_Float16)f.y;
      Bs[(c4 + 2) * 40 + kr] = (_Float16)f.z;
      Bs[(c4 + 3) * 40 + kr] = (_Float16)f.w;
    }
    __syncthreads();
    v16h a = load_a_frag(As + wave * 16 * 40, 40, lane);
#pragma unroll
    for (int t = 0; t < 4; ++t) {
      v16h b = load_b_frag(Bs + t * 16 * 40, 40, lane);
      acc[t] = wmma_f16(a, b, acc[t]);
    }
    __syncthreads();
  }

  const int rl = (lane >> 4) << 3, cl = lane & 15;
#pragma unroll
  for (int t = 0; t < 4; ++t) {
    int gc = gcBase + t * 16 + cl;                   // region uniform per block
#pragma unroll
    for (int r = 0; r < 8; ++r) {
      int m  = rowTile + wave * 16 + rl + r;
      int bb = m >> 12, n = m & (NSEQ - 1);
      float v = acc[t][r];
      if (gc < 512) {
        int h = gc >> 6, dd = gc & 63;
        qh[(((size_t)(bb * HEADS + h)) * NSEQ + n) * DHEAD + dd] = (_Float16)(v * ATTN_SCALE);
      } else if (gc < 1024) {
        int c = gc - 512, h = c >> 6, dd = c & 63;
        kh[(((size_t)(bb * HEADS + h)) * NSEQ + n) * DHEAD + dd] = (_Float16)v;
      } else {
        int c = gc - 1024, h = c >> 6, dd = c & 63;
        vh[(((size_t)(bb * HEADS + h)) * NSEQ + n) * DHEAD + dd] = (_Float16)v;
      }
    }
  }
}

// ---------------------------------------------------------------------------
// Kernel 2: flash attention. 4 waves x 16 q-rows; 64-key blocks; online
// softmax in f32 registers. K tile DMA'd to LDS by the Tensor Data Mover
// (wave 0 issues, s_wait_tensorcnt, barrier publishes); TDM row padding
// (pad_interval=32dw, pad_amount=4dw) yields the conflict-free 72-half stride.
// ---------------------------------------------------------------------------
__global__ __launch_bounds__(128) void flash_attn_kernel(
    const _Float16* __restrict__ qh, const _Float16* __restrict__ kg,
    const _Float16* __restrict__ vg, _Float16* __restrict__ ah)
{
  __shared__ __align__(16) _Float16 VT[64 * 72];      // V^T tile [d][key]
  __shared__ __align__(16) _Float16 Ps[4][16 * 72];   // per-wave P tile [row][key]
#if USE_TDM
  __shared__ __align__(16) _Float16 Ks[64 * 72];      // K tile [key][d], stride 72 via TDM pad
#endif
  const int tid = threadIdx.x, lane = tid & 31, wave = tid >> 5;
  const int bh    = blockIdx.y;                       // b*8 + h
  const int qbase = blockIdx.x * 64;
  const _Float16* Q = qh + (size_t)bh * NSEQ * DHEAD;
  const _Float16* K = kg + (size_t)bh * NSEQ * DHEAD;
  const _Float16* V = vg + (size_t)bh * NSEQ * DHEAD;

  v16h qf0, qf1;
  {
    const _Float16* Qrow = Q + (size_t)(qbase + wave * 16) * DHEAD;
    qf0 = load_a_frag(Qrow,      DHEAD, lane);        // d 0..31
    qf1 = load_a_frag(Qrow + 32, DHEAD, lane);        // d 32..63
  }

  float m_run[8], l_run[8];
  v8f o[4] = {};
#pragma unroll
  for (int r = 0; r < 8; ++r) { m_run[r] = -1e30f; l_run[r] = 0.f; }

  const int rl = (lane >> 4) << 3, cl = lane & 15;

#if USE_TDM
  // constant part of the K-tile tensor descriptor (2D, 64x64 f16 tile)
  i32x8 g1;
  g1[0] = (int)((1u << 16)            // data_size = 2 bytes
              | (1u << 20)            // pad_enable
              | (4u << 22)            // pad_interval: 32 DWORDs (one 128B row)
              | (3u << 25));          // pad_amount: 4 DWORDs (16B) -> stride 144B
  g1[1] = (int)((64u & 0xFFFFu) << 16);          // tensor_dim0 = 64 (lo16)
  g1[2] = (int)((NSEQ & 0xFFFFu) << 16);         // tensor_dim0 hi | tensor_dim1 lo = 4096
  g1[3] = (int)(64u << 16);                      // tensor_dim1 hi | tile_dim0 = 64
  g1[4] = (int)64u;                              // tile_dim1 = 64, tile_dim2 = 0
  g1[5] = (int)64u;                              // tensor_dim0_stride = 64 elems
  g1[6] = 0; g1[7] = 0;
  const i32x4 gz  = {0, 0, 0, 0};
  const i32x8 gz8 = {0, 0, 0, 0, 0, 0, 0, 0};
  const uint32_t lds_base = (uint32_t)(uintptr_t)(&Ks[0]);
#endif

  for (int j0 = 0; j0 < NSEQ; j0 += 64) {
    __syncthreads();                                  // LDS tiles free to overwrite
#if USE_TDM
    if (wave == 0) {                                  // one DMA per block
      uint64_t ga = (uint64_t)(uintptr_t)(K + (size_t)j0 * DHEAD);
      u32x4 g0;
      g0[0] = 1u;                                     // count=1, user mode
      g0[1] = lds_base;                               // LDS byte address
      g0[2] = (uint32_t)ga;                           // global_addr[31:0]
      g0[3] = (uint32_t)((ga >> 32) & 0x01FFFFFFull) | (2u << 30);  // addr[56:32] | type=2
      __builtin_amdgcn_tensor_load_to_lds(g0, g1, gz, gz, gz8, 0);
    }
#endif
    // stage V^T: V[j0+j][d] -> VT[d][j]
#pragma unroll
    for (int it = 0; it < 16; ++it) {
      int idx = tid + it * 128;
      int j  = idx >> 5;
      int d2 = (idx & 31) << 1;
      union { uint32_t u; _Float16 f[2]; } cv;
      cv.u = *(const uint32_t*)(V + (size_t)(j0 + j) * DHEAD + d2);
      VT[(d2 + 0) * 72 + j] = cv.f[0];
      VT[(d2 + 1) * 72 + j] = cv.f[1];
    }
    if (j0 + 64 < NSEQ) {
      __builtin_prefetch((const void*)(V + (size_t)(j0 + 64) * DHEAD + tid * 32), 0, 0);
    }
#if USE_TDM
    if (wave == 0) __builtin_amdgcn_s_wait_tensorcnt(0);
#endif
    __syncthreads();

    // S = Q K^T : 4 tiles of 16 keys x 2 d-chunks
    v8f s[4];
#pragma unroll
    for (int t = 0; t < 4; ++t) {
#if USE_TDM
      const _Float16* Krow = Ks + (t * 16) * 72;
      v16h b0 = load_b_frag(Krow,      72, lane);
      v16h b1 = load_b_frag(Krow + 32, 72, lane);
#else
      const _Float16* Krow = K + (size_t)(j0 + t * 16) * DHEAD;
      v16h b0 = load_b_frag(Krow,      DHEAD, lane);
      v16h b1 = load_b_frag(Krow + 32, DHEAD, lane);
#endif
      v8f z = {};
      z = wmma_f16(qf0, b0, z);
      z = wmma_f16(qf1, b1, z);
      s[t] = z;
    }

    // online softmax (row's 16 elements live in one 16-lane half)
#pragma unroll
    for (int r = 0; r < 8; ++r) {
      float mt = fmaxf(fmaxf(s[0][r], s[1][r]), fmaxf(s[2][r], s[3][r]));
#pragma unroll
      for (int off = 1; off < 16; off <<= 1) mt = fmaxf(mt, __shfl_xor(mt, off, 32));
      float mnew  = fmaxf(m_run[r], mt);
      float alpha = __expf(m_run[r] - mnew);
      float sum = 0.f;
#pragma unroll
      for (int t = 0; t < 4; ++t) {
        float p = __expf(s[t][r] - mnew);
        s[t][r] = p;
        sum += p;
      }
#pragma unroll
      for (int off = 1; off < 16; off <<= 1) sum += __shfl_xor(sum, off, 32);
      l_run[r] = l_run[r] * alpha + sum;
      m_run[r] = mnew;
#pragma unroll
      for (int t = 0; t < 4; ++t) o[t][r] *= alpha;
    }

    // P (D-layout) -> per-wave LDS -> A-layout fragments
    _Float16* Pw = &Ps[wave][0];
#pragma unroll
    for (int t = 0; t < 4; ++t)
#pragma unroll
      for (int r = 0; r < 8; ++r)
        Pw[(rl + r) * 72 + t * 16 + cl] = (_Float16)s[t][r];

    v16h pa0 = load_a_frag(Pw,      72, lane);        // keys 0..31
    v16h pa1 = load_a_frag(Pw + 32, 72, lane);        // keys 32..63

    // O += P V  (B-frags from VT[d][key])
#pragma unroll
    for (int t = 0; t < 4; ++t) {
      v16h vb0 = load_b_frag(VT + t * 16 * 72,      72, lane);
      v16h vb1 = load_b_frag(VT + t * 16 * 72 + 32, 72, lane);
      o[t] = wmma_f16(pa0, vb0, o[t]);
      o[t] = wmma_f16(pa1, vb1, o[t]);
    }
  }

  // finalize: O /= l, write f16 in [B*N, h*d]
  const int h = bh & 7, b = bh >> 3;
#pragma unroll
  for (int t = 0; t < 4; ++t)
#pragma unroll
    for (int r = 0; r < 8; ++r) {
      int n = qbase + wave * 16 + rl + r;
      float val = o[t][r] / l_run[r];
      ah[((size_t)(b * NSEQ + n)) * DMODEL + h * DHEAD + t * 16 + cl] = (_Float16)val;
    }
}

// ---------------------------------------------------------------------------
// Kernel 3: out projection: ah[8192,512]f16 @ Wo[512,512](f32->f16) + bo -> f32
// ---------------------------------------------------------------------------
__global__ __launch_bounds__(128) void out_proj_kernel(
    const _Float16* __restrict__ ah, const float* __restrict__ Wo,
    const float* __restrict__ bo, float* __restrict__ out)
{
  __shared__ __align__(16) _Float16 As[64 * 40];
  __shared__ __align__(16) _Float16 Bs[64 * 40];
  const int tid = threadIdx.x, lane = tid & 31, wave = tid >> 5;
  const int rowTile = blockIdx.x * 64;
  const int gcBase  = blockIdx.y * 64;

  v8f acc[4] = {};

  for (int k0 = 0; k0 < DMODEL; k0 += 32) {
#pragma unroll
    for (int it = 0; it < 8; ++it) {                  // 1024 u32 of f16 A data
      int idx = tid + it * 128;
      int r  = idx >> 4;
      int c2 = (idx & 15) << 1;
      uint32_t pk = *(const uint32_t*)(ah + (size_t)(rowTile + r) * DMODEL + k0 + c2);
      *(uint32_t*)(As + r * 40 + c2) = pk;
    }
#pragma unroll
    for (int it = 0; it < 4; ++it) {
      int idx = tid + it * 128;
      int kr = idx >> 4;
      int c4 = (idx & 15) << 2;
      float4 f = *(const float4*)(Wo + (size_t)(k0 + kr) * 512 + gcBase + c4);
      Bs[(c4 + 0) * 40 + kr] = (_Float16)f.x;
      Bs[(c4 + 1) * 40 + kr] = (_Float16)f.y;
      Bs[(c4 + 2) * 40 + kr] = (_Float16)f.z;
      Bs[(c4 + 3) * 40 + kr] = (_Float16)f.w;
    }
    __syncthreads();
    v16h a = load_a_frag(As + wave * 16 * 40, 40, lane);
#pragma unroll
    for (int t = 0; t < 4; ++t) {
      v16h b = load_b_frag(Bs + t * 16 * 40, 40, lane);
      acc[t] = wmma_f16(a, b, acc[t]);
    }
    __syncthreads();
  }

  const int rl = (lane >> 4) << 3, cl = lane & 15;
#pragma unroll
  for (int t = 0; t < 4; ++t) {
    int gc = gcBase + t * 16 + cl;
    float bias = bo[gc];
#pragma unroll
    for (int r = 0; r < 8; ++r) {
      int m = rowTile + wave * 16 + rl + r;
      out[(size_t)m * DMODEL + gc] = acc[t][r] + bias;
    }
  }
}

// ---------------------------------------------------------------------------
extern "C" void kernel_launch(void* const* d_in, const int* in_sizes, int n_in,
                              void* d_out, int out_size, void* d_ws, size_t ws_size,
                              hipStream_t stream) {
  const float* x   = (const float*)d_in[0];
  const float* Wq  = (const float*)d_in[1];
  const float* Wkv = (const float*)d_in[2];
  const float* Wo  = (const float*)d_in[3];
  const float* bo  = (const float*)d_in[4];
  float* out = (float*)d_out;

  // workspace: q,k,v,attn-out f16, each 2*8*4096*64 = 4,194,304 halves (8MB)
  const size_t perT = (size_t)2 * HEADS * NSEQ * DHEAD;
  _Float16* qh = (_Float16*)d_ws;
  _Float16* kh = qh + perT;
  _Float16* vh = kh + perT;
  _Float16* ah = vh + perT;

  dim3 blk(128);
  dim3 g1(8192 / 64, 1536 / 64);          // 128 x 24
  qkv_proj_kernel<<<g1, blk, 0, stream>>>(x, Wq, Wkv, qh, kh, vh);

  dim3 g2(NSEQ / 64, 2 * HEADS);          // 64 x 16
  flash_attn_kernel<<<g2, blk, 0, stream>>>(qh, kh, vh, ah);

  dim3 g3(8192 / 64, DMODEL / 64);        // 128 x 8
  out_proj_kernel<<<g3, blk, 0, stream>>>(ah, Wo, bo, out);
}
